// CSPVLayer_33629593927945
// MI455X (gfx1250) — matrix operand order repaired and checked by reading, not compiled, not run on hardware
//
#include <hip/hip_runtime.h>
#include <hip/hip_bf16.h>

typedef __attribute__((ext_vector_type(16))) _Float16 v16h;
typedef __attribute__((ext_vector_type(8)))  _Float16 v8h;
typedef __attribute__((ext_vector_type(8)))  float    v8f;

#define N_NODES 50000
#define N_EDGES 400000
#define HDIM    128
#define IN_DIM  518
#define KPAD    544          // 518 padded to 17 * 32
#define KT_E1   17           // k-tiles for 544-deep GEMM
#define KT_128  4            // k-tiles for 128-deep GEMM
#define KT_256  8            // k-tiles for 256-deep GEMM
#define NTILES  8            // 128 / 16 output column tiles
#define MSUB    2            // two 16-row A sub-tiles per wave (M = 32)

union V16 { v16h v; v8h h[2]; };

__device__ __forceinline__ float silu_f(float x) { return x / (1.0f + __expf(-x)); }

// A-fragment loader for v_wmma_f32_16x16x32_f16 from a row-major f16 LDS tile.
// Per ISA 16-bit A layout: lane L (row = L&15) holds K = kb..kb+7 and kb+16..kb+23
// where kb = ktBase + (L>>4)*8.  Two ds_load_b128 per fragment.
__device__ __forceinline__ v16h load_a_frag(const _Float16* row0, int rowStrideHalves,
                                            int ktBase, int lane) {
  const int row = lane & 15;
  const int kb  = ktBase + ((lane >> 4) << 3);
  const _Float16* p = row0 + row * rowStrideHalves + kb;
  V16 u;
  u.h[0] = *(const v8h*)(p);
  u.h[1] = *(const v8h*)(p + 16);
  return u.v;
}

// ---------------------------------------------------------------------------
// Kernel 0: zero scatter accumulator (d_out doubles as sums) + counts
// ---------------------------------------------------------------------------
__global__ void zero_buffers(float* __restrict__ sums, float* __restrict__ counts) {
  const int tid    = blockIdx.x * blockDim.x + threadIdx.x;
  const int stride = gridDim.x * blockDim.x;
  for (int i = tid; i < N_NODES * HDIM; i += stride) sums[i] = 0.0f;
  for (int i = tid; i < N_NODES; i += stride) counts[i] = 0.0f;
}

// ---------------------------------------------------------------------------
// Kernel 1: repack fp32 weights into pre-swizzled f16 WMMA B-fragments.
// B layout (32x16 f16): lane half h holds K = 8h..8h+7 (i<8) and 16+8h..16+8h+7 (i>=8),
// column = lane&15, matching the A fragment K order.  One fragment = 512 halves.
// ---------------------------------------------------------------------------
__device__ __forceinline__ void pack_frags(const float* __restrict__ W,
                                           _Float16* __restrict__ F,
                                           int ktiles, int klimit,
                                           int tid, int stride) {
  const int total = ktiles * NTILES * 512;
  for (int idx = tid; idx < total; idx += stride) {
    const int elem = idx & 511;
    const int frag = idx >> 9;
    const int lane = elem >> 4;
    const int i    = elem & 15;
    const int kt   = frag >> 3;
    const int nt   = frag & 7;
    const int k    = kt * 32 + ((lane >> 4) << 3) + (i < 8 ? i : 8 + i);
    const int n    = nt * 16 + (lane & 15);
    F[idx] = (_Float16)((k < klimit) ? W[k * HDIM + n] : 0.0f);
  }
}

__global__ void prep_weights(const float* __restrict__ e_w1, const float* __restrict__ e_w2,
                             const float* __restrict__ n_w1, const float* __restrict__ n_w2,
                             _Float16* __restrict__ w1f, _Float16* __restrict__ w2f,
                             _Float16* __restrict__ m1f, _Float16* __restrict__ m2f) {
  const int tid    = blockIdx.x * blockDim.x + threadIdx.x;
  const int stride = gridDim.x * blockDim.x;
  pack_frags(e_w1, w1f, KT_E1, IN_DIM, tid, stride);
  pack_frags(e_w2, w2f, KT_128, 128,  tid, stride);
  pack_frags(n_w1, m1f, KT_256, 256,  tid, stride);
  pack_frags(n_w2, m2f, KT_128, 128,  tid, stride);
}

// ---------------------------------------------------------------------------
// Kernel 2: fused edge MLP + scatter-add.  One wave (one block of 32) owns
// 32 edges = two 16-row WMMA sub-tiles that share every B fragment.
// ---------------------------------------------------------------------------
__global__ __launch_bounds__(32) void edge_mlp_scatter(
    const float* __restrict__ pos_diff, const float* __restrict__ v3,
    const float* __restrict__ h,        const float* __restrict__ lmat,
    const int*   __restrict__ eni,      const int*   __restrict__ egi,
    const float* __restrict__ vw,       const float* __restrict__ vb,
    const float* __restrict__ eb1,      const float* __restrict__ eb2,
    const _Float16* __restrict__ w1f,   const _Float16* __restrict__ w2f,
    float* __restrict__ sums,           float* __restrict__ counts) {
  __shared__ _Float16 s_in[32 * KPAD];   // 34816 B
  __shared__ _Float16 s_ef[32 * HDIM];   //  8192 B

  const int lane = threadIdx.x & 31;
  const int base = blockIdx.x * 32;

  // ---- stage edges_input = [hi | hj | l | vij | pd | 0-pad] as f16 ----
  for (int e = 0; e < 32; ++e) {
    const int edge = base + e;
    const int s = eni[edge];
    const int d = eni[N_EDGES + edge];
    const int g = egi[edge];
    const float dx = v3[d * 3 + 0] - v3[s * 3 + 0];
    const float dy = v3[d * 3 + 1] - v3[s * 3 + 1];
    const float dz = v3[d * 3 + 2] - v3[s * 3 + 2];
    const float pdv = pos_diff[edge];
    _Float16* rowp = s_in + e * KPAD;
    for (int c = lane; c < HDIM; c += 32) {
      rowp[c]       = (_Float16)h[(size_t)s * HDIM + c];
      rowp[128 + c] = (_Float16)h[(size_t)d * HDIM + c];
      rowp[262 + c] = (_Float16)(dx * vw[c] + dy * vw[128 + c] + dz * vw[256 + c] + vb[c]);
    }
    if (lane < 6)  rowp[256 + lane] = (_Float16)lmat[g * 6 + lane];
    if (lane < 26) rowp[518 + lane] = (_Float16)0.0f;
    for (int c = lane; c < 64; c += 32) {
      const float fr  = __expf(-9.2103403719761836f * (float)c * (1.0f / 64.0f));
      const float ang = pdv * fr;
      rowp[390 + c]      = (_Float16)__sinf(ang);
      rowp[390 + 64 + c] = (_Float16)__cosf(ang);
    }
  }
  // single wave: DS ops are in-order -> no barrier needed

  const int colBase = lane & 15;
  const int rowOfs  = (lane >> 4) << 3;

  v8f acc[MSUB][NTILES];
  const v8f vzero = {0.f, 0.f, 0.f, 0.f, 0.f, 0.f, 0.f, 0.f};
#pragma unroll
  for (int m = 0; m < MSUB; ++m)
#pragma unroll
    for (int nt = 0; nt < NTILES; ++nt) acc[m][nt] = vzero;

  // ---- GEMM1: [32,544] x [544,128]; each B fragment feeds 2 WMMAs ----
  for (int kt = 0; kt < KT_E1; ++kt) {
    v16h a[MSUB];
#pragma unroll
    for (int m = 0; m < MSUB; ++m)
      a[m] = load_a_frag(s_in + m * 16 * KPAD, KPAD, kt * 32, lane);
    const _Float16* bf = w1f + (size_t)(kt * NTILES) * 512 + lane * 16;
    if (kt + 1 < KT_E1)
      __builtin_prefetch((const void*)(bf + NTILES * 512), 0, 3);
#pragma unroll
    for (int nt = 0; nt < NTILES; ++nt) {
      const v16h b = *(const v16h*)(bf + nt * 512);
#pragma unroll
      for (int m = 0; m < MSUB; ++m)
        acc[m][nt] = __builtin_amdgcn_wmma_f32_16x16x32_f16(false, a[m], false, b,
                                                            (short)0, acc[m][nt],
                                                            false, false);
    }
  }

  // bias + SiLU -> ef tile (f16) in LDS
#pragma unroll
  for (int nt = 0; nt < NTILES; ++nt) {
    const float bia = eb1[nt * 16 + colBase];
#pragma unroll
    for (int m = 0; m < MSUB; ++m) {
#pragma unroll
      for (int r = 0; r < 8; ++r) {
        const float x = acc[m][nt][r] + bia;
        s_ef[(m * 16 + r + rowOfs) * HDIM + nt * 16 + colBase] = (_Float16)silu_f(x);
      }
      acc[m][nt] = vzero;
    }
  }

  // ---- GEMM2: [32,128] x [128,128] ----
  for (int kt = 0; kt < KT_128; ++kt) {
    v16h a[MSUB];
#pragma unroll
    for (int m = 0; m < MSUB; ++m)
      a[m] = load_a_frag(s_ef + m * 16 * HDIM, HDIM, kt * 32, lane);
    const _Float16* bf = w2f + (size_t)(kt * NTILES) * 512 + lane * 16;
#pragma unroll
    for (int nt = 0; nt < NTILES; ++nt) {
      const v16h b = *(const v16h*)(bf + nt * 512);
#pragma unroll
      for (int m = 0; m < MSUB; ++m)
        acc[m][nt] = __builtin_amdgcn_wmma_f32_16x16x32_f16(false, a[m], false, b,
                                                            (short)0, acc[m][nt],
                                                            false, false);
    }
  }

  // bias + SiLU + scatter-add onto source nodes
  int srow[MSUB][8];
#pragma unroll
  for (int m = 0; m < MSUB; ++m)
#pragma unroll
    for (int r = 0; r < 8; ++r) srow[m][r] = eni[base + m * 16 + r + rowOfs];

#pragma unroll
  for (int nt = 0; nt < NTILES; ++nt) {
    const float bia = eb2[nt * 16 + colBase];
#pragma unroll
    for (int m = 0; m < MSUB; ++m)
#pragma unroll
      for (int r = 0; r < 8; ++r) {
        const float val = silu_f(acc[m][nt][r] + bia);
        atomicAdd(&sums[(size_t)srow[m][r] * HDIM + nt * 16 + colBase], val);
      }
  }
  if (lane < 16) {
    atomicAdd(&counts[eni[base + lane]], 1.0f);
    atomicAdd(&counts[eni[base + 16 + lane]], 1.0f);
  }
}

// ---------------------------------------------------------------------------
// Kernel 3: node MLP + residual.  One wave owns 32 nodes (two 16-row
// sub-tiles).  Reads sums (agg) out of d_out, then overwrites d_out.
// 50000 % 32 != 0: staging clamps reads (stays inside this wave's own rows),
// stores are guarded per element (after all WMMAs -> EXEC full for matrix ops).
// ---------------------------------------------------------------------------
__global__ __launch_bounds__(32) void node_mlp(
    const float* __restrict__ h,
    const float* __restrict__ nb1, const float* __restrict__ nb2,
    const _Float16* __restrict__ m1f, const _Float16* __restrict__ m2f,
    const float* __restrict__ counts,
    float* __restrict__ out_sums) {
  __shared__ _Float16 s_in[32 * 256];   // 16384 B
  __shared__ _Float16 s_ef[32 * HDIM];  //  8192 B

  const int lane = threadIdx.x & 31;
  const int base = blockIdx.x * 32;

  // stage nin = [h | agg] as f16 (clamped reads for the ragged last tile)
  for (int e = 0; e < 32; ++e) {
    const int node = min(base + e, N_NODES - 1);
    const float inv = 1.0f / fmaxf(counts[node], 1.0f);
    for (int c = lane; c < HDIM; c += 32) {
      s_in[e * 256 + c]       = (_Float16)h[(size_t)node * HDIM + c];
      s_in[e * 256 + 128 + c] = (_Float16)(out_sums[(size_t)node * HDIM + c] * inv);
    }
  }

  const int colBase = lane & 15;
  const int rowOfs  = (lane >> 4) << 3;

  v8f acc[MSUB][NTILES];
  const v8f vzero = {0.f, 0.f, 0.f, 0.f, 0.f, 0.f, 0.f, 0.f};
#pragma unroll
  for (int m = 0; m < MSUB; ++m)
#pragma unroll
    for (int nt = 0; nt < NTILES; ++nt) acc[m][nt] = vzero;

  // GEMM1: [32,256] x [256,128]
  for (int kt = 0; kt < KT_256; ++kt) {
    v16h a[MSUB];
#pragma unroll
    for (int m = 0; m < MSUB; ++m)
      a[m] = load_a_frag(s_in + m * 16 * 256, 256, kt * 32, lane);
    const _Float16* bf = m1f + (size_t)(kt * NTILES) * 512 + lane * 16;
    if (kt + 1 < KT_256)
      __builtin_prefetch((const void*)(bf + NTILES * 512), 0, 3);
#pragma unroll
    for (int nt = 0; nt < NTILES; ++nt) {
      const v16h b = *(const v16h*)(bf + nt * 512);
#pragma unroll
      for (int m = 0; m < MSUB; ++m)
        acc[m][nt] = __builtin_amdgcn_wmma_f32_16x16x32_f16(false, a[m], false, b,
                                                            (short)0, acc[m][nt],
                                                            false, false);
    }
  }

#pragma unroll
  for (int nt = 0; nt < NTILES; ++nt) {
    const float bia = nb1[nt * 16 + colBase];
#pragma unroll
    for (int m = 0; m < MSUB; ++m) {
#pragma unroll
      for (int r = 0; r < 8; ++r) {
        const float x = acc[m][nt][r] + bia;
        s_ef[(m * 16 + r + rowOfs) * HDIM + nt * 16 + colBase] = (_Float16)silu_f(x);
      }
      acc[m][nt] = vzero;
    }
  }

  // GEMM2: [32,128] x [128,128]
  for (int kt = 0; kt < KT_128; ++kt) {
    v16h a[MSUB];
#pragma unroll
    for (int m = 0; m < MSUB; ++m)
      a[m] = load_a_frag(s_ef + m * 16 * HDIM, HDIM, kt * 32, lane);
    const _Float16* bf = m2f + (size_t)(kt * NTILES) * 512 + lane * 16;
#pragma unroll
    for (int nt = 0; nt < NTILES; ++nt) {
      const v16h b = *(const v16h*)(bf + nt * 512);
#pragma unroll
      for (int m = 0; m < MSUB; ++m)
        acc[m][nt] = __builtin_amdgcn_wmma_f32_16x16x32_f16(false, a[m], false, b,
                                                            (short)0, acc[m][nt],
                                                            false, false);
    }
  }

  // bias + SiLU + residual, overwrite d_out (guard the ragged tail)
#pragma unroll
  for (int nt = 0; nt < NTILES; ++nt) {
    const float bia = nb2[nt * 16 + colBase];
#pragma unroll
    for (int m = 0; m < MSUB; ++m)
#pragma unroll
      for (int r = 0; r < 8; ++r) {
        const int node = base + m * 16 + r + rowOfs;
        if (node < N_NODES) {
          const int col = nt * 16 + colBase;
          const float val = silu_f(acc[m][nt][r] + bia);
          out_sums[(size_t)node * HDIM + col] = h[(size_t)node * HDIM + col] + val;
        }
      }
  }
}

// ---------------------------------------------------------------------------
extern "C" void kernel_launch(void* const* d_in, const int* in_sizes, int n_in,
                              void* d_out, int out_size, void* d_ws, size_t ws_size,
                              hipStream_t stream) {
  const float* pos_diff = (const float*)d_in[0];
  const float* v3       = (const float*)d_in[1];
  const float* h        = (const float*)d_in[2];
  const float* lmat     = (const float*)d_in[3];
  const int*   eni      = (const int*)d_in[4];
  const int*   egi      = (const int*)d_in[5];
  const float* vw       = (const float*)d_in[6];
  const float* vb       = (const float*)d_in[7];
  const float* e_w1     = (const float*)d_in[8];
  const float* e_b1     = (const float*)d_in[9];
  const float* e_w2     = (const float*)d_in[10];
  const float* e_b2     = (const float*)d_in[11];
  const float* n_w1     = (const float*)d_in[12];
  const float* n_b1     = (const float*)d_in[13];
  const float* n_w2     = (const float*)d_in[14];
  const float* n_b2     = (const float*)d_in[15];
  float* out = (float*)d_out;

  // workspace layout (all offsets 32B-aligned; total ~470 KB)
  char* ws = (char*)d_ws;
  float*    counts = (float*)ws;                                        // 200000 B
  _Float16* w1f = (_Float16*)(ws + 200192);                             // 139264 B
  _Float16* w2f = (_Float16*)(ws + 200192 + 139264);                    //  32768 B
  _Float16* m1f = (_Float16*)(ws + 200192 + 139264 + 32768);            //  65536 B
  _Float16* m2f = (_Float16*)(ws + 200192 + 139264 + 32768 + 65536);    //  32768 B

  zero_buffers<<<4096, 256, 0, stream>>>(out, counts);
  prep_weights<<<512, 256, 0, stream>>>(e_w1, e_w2, n_w1, n_w2, w1f, w2f, m1f, m2f);
  edge_mlp_scatter<<<N_EDGES / 32, 32, 0, stream>>>(pos_diff, v3, h, lmat, eni, egi,
                                                    vw, vb, e_b1, e_b2, w1f, w2f,
                                                    out, counts);
  node_mlp<<<(N_NODES + 31) / 32, 32, 0, stream>>>(h, n_b1, n_b2, m1f, m2f,
                                                   counts, out);
}
